// MOSLoss_77000173683134
// MI455X (gfx1250) — compile-verified, matrix operand order checked by reading, not run on hardware
//
#include <hip/hip_runtime.h>
#include <hip/hip_bf16.h>

typedef __attribute__((ext_vector_type(16))) __bf16 v16bf;
typedef __attribute__((ext_vector_type(8)))  float  v8f;

constexpr int B_ = 128, FEAT = 512, E_ = 8, L_ = 256, V_ = 50257, T_ = 16;
constexpr int M_ = B_ * E_;               // 1024 rows of the big GEMM
constexpr int NW = 64;                    // N-chunk waves per M-tile
constexpr int MTILES = M_ / 16;           // 64
constexpr int NTILES = (V_ + 15) / 16;    // 3142 column tiles
constexpr int NPAIRS = (NTILES + 1) / 2;  // 1571 tile-pairs (32 columns each)

struct BPair { uint4 lo, hi; };

static __device__ __forceinline__ v16bf load_bf16x16(const __bf16* base, int off0, int off1) {
  BPair p;
  p.lo = *reinterpret_cast<const uint4*>(base + off0);
  p.hi = *reinterpret_cast<const uint4*>(base + off1);
  return __builtin_bit_cast(v16bf, p);
}

// ---------------- fp32 -> bf16 conversion of W_lab (one-time, streamed) ----------------
__global__ void k_convert_wlab(const float* __restrict__ w, __bf16* __restrict__ o, int n) {
  int i = blockIdx.x * blockDim.x + threadIdx.x;
  int stride = gridDim.x * blockDim.x;
  for (; i < n; i += stride) o[i] = (__bf16)w[i];
}

// ---------------- expert_h = features @ W_exp.T + b_exp  (fp32, tiny GEMM) -------------
__global__ void k_expert_h(const float* __restrict__ feat, const float* __restrict__ Wexp,
                           const float* __restrict__ bexp, __bf16* __restrict__ eh) {
  int idx = blockIdx.x * blockDim.x + threadIdx.x;   // 0 .. 128*2048-1
  int b = idx >> 11, j = idx & 2047;
  const float* f = feat + (size_t)b * FEAT;
  const float* w = Wexp + (size_t)j * FEAT;
  float acc = bexp[j];
  #pragma unroll 8
  for (int k = 0; k < FEAT; ++k) acc = fmaf(f[k], w[k], acc);
  eh[idx] = (__bf16)acc;   // [B, E*L] row-major == [B*E, L] row-major
}

// ---------------- prior = softmax(features @ W_pri.T + b_pri) --------------------------
__global__ void k_prior(const float* __restrict__ feat, const float* __restrict__ Wpri,
                        const float* __restrict__ bpri, float* __restrict__ prior) {
  int b = blockIdx.x * blockDim.x + threadIdx.x;
  if (b >= B_) return;
  const float* f = feat + (size_t)b * FEAT;
  float lg[E_];
  float mx = -1e30f;
  for (int e = 0; e < E_; ++e) {
    const float* w = Wpri + (size_t)e * FEAT;
    float acc = bpri[e];
    #pragma unroll 8
    for (int k = 0; k < FEAT; ++k) acc = fmaf(f[k], w[k], acc);
    lg[e] = acc;
    mx = fmaxf(mx, acc);
  }
  float s = 0.f;
  for (int e = 0; e < E_; ++e) { lg[e] = __expf(lg[e] - mx); s += lg[e]; }
  float inv = 1.f / s;
  for (int e = 0; e < E_; ++e) prior[b * E_ + e] = lg[e] * inv;
}

// ---------------- target logits (same bf16 numerics as the WMMA GEMM) ------------------
__global__ void k_tgt_logits(const __bf16* __restrict__ eh, const __bf16* __restrict__ wlab,
                             const float* __restrict__ blab, const int* __restrict__ targets,
                             float* __restrict__ tlog) {
  int idx = blockIdx.x * blockDim.x + threadIdx.x;   // B*T*E = 16384
  int b = idx >> 7, r = idx & 127, t = r >> 3, e = r & 7;
  int v = targets[b * T_ + t];
  const __bf16* h  = eh + (size_t)(b * E_ + e) * L_;
  const __bf16* wv = wlab + (size_t)v * L_;
  float acc = blab[v];
  #pragma unroll 8
  for (int k = 0; k < L_; ++k) acc = fmaf((float)h[k], (float)wv[k], acc);
  tlog[idx] = acc;   // [(b*T + t)*E + e]
}

// ---------------- main kernel: dual-chain WMMA logits + online softmax partials --------
__global__ void __launch_bounds__(32)
k_softmax_partials(const __bf16* __restrict__ eh, const __bf16* __restrict__ wlab,
                   const float* __restrict__ blab, float2* __restrict__ part) {
  const int lane = threadIdx.x;
  const int n  = lane & 15;        // N (or M-row) index within tile
  const int hi = lane >> 4;        // which K-chunk half this lane holds
  const int wchunk = blockIdx.x;   // 0..NW-1   (column range)
  const int mtile  = blockIdx.y;   // 0..MTILES-1

  // Preload A tile: 16 rows x 256 K of bf16, resident in 64 VGPRs/lane.
  const __bf16* arow = eh + (size_t)(mtile * 16 + n) * L_;
  v16bf A[8];
  #pragma unroll
  for (int ks = 0; ks < 8; ++ks) {
    int kb = ks * 32 + hi * 8;
    A[ks] = load_bf16x16(arow, kb, kb + 16);
  }

  // 8 online-softmax accumulators per lane (one per C-layout row slot).
  float m[8], s[8];
  #pragma unroll
  for (int j = 0; j < 8; ++j) { m[j] = -1e30f; s[j] = 0.f; }

  const int ppw = (NPAIRS + NW - 1) / NW;               // pairs per wave
  const int p0 = wchunk * ppw;
  const int p1 = (p0 + ppw < NPAIRS) ? (p0 + ppw) : NPAIRS;

  for (int pp = p0; pp < p1; ++pp) {
    const int t  = pp * 2;                              // two 16-col tiles per iteration
    const int v0 = t * 16 + n;
    const int v1 = v0 + 16;
    const int vc0 = (v0 < V_) ? v0 : (V_ - 1);          // clamp loads, mask later
    const int vc1 = (v1 < V_) ? v1 : (V_ - 1);
    const __bf16* br0 = wlab + (size_t)vc0 * L_;
    const __bf16* br1 = wlab + (size_t)vc1 * L_;
    __builtin_prefetch(br0 + 32 * L_, 0, 1);            // next pair's columns
    __builtin_prefetch(br1 + 32 * L_, 0, 1);

    v8f c0 = {}, c1 = {};
    #pragma unroll
    for (int ks = 0; ks < 8; ++ks) {                    // two independent D->C chains
      int kb = ks * 32 + hi * 8;
      v16bf B0 = load_bf16x16(br0, kb, kb + 16);
      v16bf B1 = load_bf16x16(br1, kb, kb + 16);
      c0 = __builtin_amdgcn_wmma_f32_16x16x32_bf16(false, A[ks], false, B0,
                                                   (short)0, c0, false, false);
      c1 = __builtin_amdgcn_wmma_f32_16x16x32_bf16(false, A[ks], false, B1,
                                                   (short)0, c1, false, false);
    }

    const float bias0 = blab[vc0];
    const float bias1 = blab[vc1];
    const bool va0 = v0 < V_;
    const bool va1 = v1 < V_;
    #pragma unroll
    for (int j = 0; j < 8; ++j) {                       // merged update: 3 exps / 2 cols
      float x0 = va0 ? (c0[j] + bias0) : -3e38f;
      float x1 = va1 ? (c1[j] + bias1) : -3e38f;
      float mn = fmaxf(m[j], fmaxf(x0, x1));
      s[j] = s[j] * __expf(m[j] - mn) + __expf(x0 - mn) + __expf(x1 - mn);
      m[j] = mn;
    }
  }

  // Merge the 16 column-partials per row (xor<16 keeps the two halves separate).
  #pragma unroll
  for (int off = 8; off >= 1; off >>= 1) {
    #pragma unroll
    for (int j = 0; j < 8; ++j) {
      float mo = __shfl_xor(m[j], off, 32);
      float so = __shfl_xor(s[j], off, 32);
      float mn = fmaxf(m[j], mo);
      s[j] = s[j] * __expf(m[j] - mn) + so * __expf(mo - mn);
      m[j] = mn;
    }
  }
  if (n == 0) {
    #pragma unroll
    for (int j = 0; j < 8; ++j) {
      int row = mtile * 16 + j + hi * 8;   // C layout: VGPR j -> M=j (lanes<16), M=j+8 (lanes>=16)
      part[(size_t)row * NW + wchunk] = make_float2(m[j], s[j]);
    }
  }
}

// ---------------- combine partials -> logZ per row -------------------------------------
__global__ void k_combine(const float2* __restrict__ part, float* __restrict__ logZ) {
  int r = blockIdx.x * blockDim.x + threadIdx.x;
  if (r >= M_) return;
  float M = -1e30f, S = 0.f;
  for (int w = 0; w < NW; ++w) {
    float2 p = part[(size_t)r * NW + w];
    float mn = fmaxf(M, p.x);
    S = S * __expf(M - mn) + p.y * __expf(p.x - mn);
    M = mn;
  }
  logZ[r] = M + __logf(S);
}

// ---------------- final weighted-NLL reduction ------------------------------------------
__global__ void k_loss(const float* __restrict__ tlog, const float* __restrict__ logZ,
                       const float* __restrict__ prior, const int* __restrict__ targets,
                       const float* __restrict__ discard, float* __restrict__ out) {
  __shared__ float red[B_];
  int b = threadIdx.x;
  float num = 0.f, den = 0.f;
  for (int t = 0; t < T_; ++t) {
    int v = targets[b * T_ + t];
    float w = 1.f - discard[v];
    float mix = 0.f;
    #pragma unroll
    for (int e = 0; e < E_; ++e)
      mix += prior[b * E_ + e] * __expf(tlog[(b * T_ + t) * E_ + e] - logZ[b * E_ + e]);
    num += -__logf(mix) * w;
    den += w;
  }
  red[b] = num / den;
  __syncthreads();
  for (int o = 64; o >= 1; o >>= 1) {
    if (b < o) red[b] += red[b + o];
    __syncthreads();
  }
  if (b == 0) out[0] = red[0] / ((float)B_ + 1e-5f);
}

extern "C" void kernel_launch(void* const* d_in, const int* in_sizes, int n_in,
                              void* d_out, int out_size, void* d_ws, size_t ws_size,
                              hipStream_t stream) {
  (void)in_sizes; (void)n_in; (void)out_size; (void)ws_size;
  const float* features = (const float*)d_in[0];
  const float* W_exp    = (const float*)d_in[1];
  const float* b_exp    = (const float*)d_in[2];
  const float* W_lab    = (const float*)d_in[3];
  const float* b_lab    = (const float*)d_in[4];
  const float* W_pri    = (const float*)d_in[5];
  const float* b_pri    = (const float*)d_in[6];
  const float* discard  = (const float*)d_in[7];
  const int*   targets  = (const int*)d_in[8];
  float* out = (float*)d_out;

  char* p = (char*)d_ws;
  auto alloc = [&](size_t bytes) { char* r = p; p += (bytes + 255) & ~(size_t)255; return r; };
  __bf16*  wlab  = (__bf16*)alloc((size_t)V_ * L_ * 2);     // 25.7 MB
  __bf16*  eh    = (__bf16*)alloc((size_t)M_ * L_ * 2);     // 0.5 MB
  float*   prior = (float*)alloc((size_t)B_ * E_ * 4);
  float*   tlog  = (float*)alloc((size_t)B_ * T_ * E_ * 4);
  float2*  part  = (float2*)alloc((size_t)M_ * NW * 8);     // 0.5 MB
  float*   logZ  = (float*)alloc((size_t)M_ * 4);

  k_convert_wlab<<<4096, 256, 0, stream>>>(W_lab, wlab, V_ * L_);
  k_expert_h<<<(B_ * E_ * L_) / 256, 256, 0, stream>>>(features, W_exp, b_exp, eh);
  k_prior<<<1, 128, 0, stream>>>(features, W_pri, b_pri, prior);
  k_tgt_logits<<<(B_ * T_ * E_) / 256, 256, 0, stream>>>(eh, wlab, b_lab, targets, tlog);
  k_softmax_partials<<<dim3(NW, MTILES), 32, 0, stream>>>(eh, wlab, b_lab, part);
  k_combine<<<(M_ + 255) / 256, 256, 0, stream>>>(part, logZ);
  k_loss<<<1, B_, 0, stream>>>(tlog, logZ, prior, targets, discard, out);
}
